// NCF_81870666596358
// MI455X (gfx1250) — matrix operand order repaired and checked by reading, not compiled, not run on hardware
//
#include <hip/hip_runtime.h>
#include <hip/hip_bf16.h>
#include <math.h>

// ---------------- problem sizes ----------------
#define U_N   6000
#define I_N   12000
#define B_N   8192
#define D_N   64
#define DM_N  256

typedef __attribute__((ext_vector_type(16))) __bf16          v16bf;
typedef __attribute__((ext_vector_type(8)))  float           v8f;
typedef unsigned short                                        ushort_t;
typedef __attribute__((ext_vector_type(16))) unsigned short  v16u;

// A-operand K mapping for V_WMMA_*_16X16X32 (ISA 7.12.2 "16-bit A-Matrix 16x32"):
// lanes 0-15: e0..7 -> K 0..7,  e8..15 -> K 16..23
// lanes16-31: e0..7 -> K 8..15, e8..15 -> K 24..31
__device__ __forceinline__ int kmapA(int e, int half) {
  return (e & 7) | (half << 3) | ((e >> 3) << 4);
}
// B-operand K mapping: same as A (lanes 0-15: K 0..7,16..23; lanes 16-31: K 8..15,24..31)
__device__ __forceinline__ int kmapB(int e, int half) { return kmapA(e, half); }

__device__ __forceinline__ v16bf ld16g(const ushort_t* p) {  // 32B aligned
  return __builtin_bit_cast(v16bf, *(const v16u*)p);
}

// f32 p[0..7], p[16..23] -> bf16 B-tile fragment (native cvt)
__device__ __forceinline__ v16bf ldrow_bf16(const float* p) {
  union { __bf16 h[16]; v16bf v; } o;
#pragma unroll
  for (int e = 0; e < 16; ++e) o.h[e] = (__bf16)p[e + (e >> 3) * 8];
  return o.v;
}

__device__ __forceinline__ v8f wmma_bf16(v16bf a, v16bf b, v8f c) {
  return __builtin_amdgcn_wmma_f32_16x16x32_bf16(false, a, false, b, (short)0, c, false, false);
}

__device__ __forceinline__ unsigned pack2bf(float a, float b) {
  union { __bf16 h[2]; unsigned u; } t;
  t.h[0] = (__bf16)a; t.h[1] = (__bf16)b;
  return t.u;
}

// relu(acc+bias) on a chunk pair, kept per half so the result is a
// valid B-operand tile (K = feature) for the next layer. Chunk0 covers
// features c0base(+r), chunk1 covers c0base+16(+r), r = 0..7 on this half.
__device__ __forceinline__ v16bf fuse_pair(const v8f& a0, const v8f& a1,
                                           const float* __restrict__ bias,
                                           int c0base, int half) {
  unsigned set0[4], set1[4];
#pragma unroll
  for (int d = 0; d < 4; ++d) {
    set0[d] = pack2bf(fmaxf(a0[2*d]   + bias[c0base + 2*d],      0.f),
                      fmaxf(a0[2*d+1] + bias[c0base + 2*d + 1],  0.f));
    set1[d] = pack2bf(fmaxf(a1[2*d]   + bias[c0base + 16 + 2*d],     0.f),
                      fmaxf(a1[2*d+1] + bias[c0base + 16 + 2*d + 1], 0.f));
  }
  union { unsigned d[8]; v16bf v; } o;
#pragma unroll
  for (int d = 0; d < 4; ++d) {
    o.d[d]     = set0[d];
    o.d[4 + d] = set1[d];
  }
  return o.v;
}

// ---------------- pack kernels (cold) ----------------

// Row-major [K,N] f32 W -> A-layout tiles of W^T:  [kstep][nchunk][lane][16]
__global__ void ncf_packAT(const float* __restrict__ src, ushort_t* __restrict__ dst,
                           int K, int N) {
  int lane = threadIdx.x, half = lane >> 4, li = lane & 15;
  int nch = N >> 4;
  int ks = blockIdx.x / nch, nc = blockIdx.x % nch;
  int n = nc * 16 + li;
  union { __bf16 h[16]; v16u v; } o;
#pragma unroll
  for (int e = 0; e < 16; ++e) {
    int k = ks * 32 + kmapA(e, half);
    o.h[e] = (__bf16)src[(size_t)k * N + n];
  }
  *(v16u*)(dst + ((size_t)blockIdx.x * 32 + lane) * 16) = o.v;
}

// Row-major [K,N] f32 -> true B-layout tiles [kstep][nchunk][lane][16], zero-pad K
__global__ void ncf_packBop(const float* __restrict__ src, ushort_t* __restrict__ dst,
                            int K, int N) {
  int lane = threadIdx.x, half = lane >> 4, li = lane & 15;
  int nch = N >> 4;
  int ks = blockIdx.x / nch, nc = blockIdx.x % nch;
  int n = nc * 16 + li;
  union { __bf16 h[16]; v16u v; } o;
#pragma unroll
  for (int e = 0; e < 16; ++e) {
    int k = ks * 32 + kmapB(e, half);
    o.h[e] = (k < K) ? (__bf16)src[(size_t)k * N + n] : (__bf16)0.f;
  }
  *(v16u*)(dst + ((size_t)blockIdx.x * 32 + lane) * 16) = o.v;
}

// Row-major [N,K] f32 (e.g. item_rel) -> B-layout tiles of src^T: [ntile][ktile][lane][16]
__global__ void ncf_packBT(const float* __restrict__ src, ushort_t* __restrict__ dst,
                           int N, int K) {
  int lane = threadIdx.x, half = lane >> 4, li = lane & 15;
  int ktiles = K >> 5;
  int nt = blockIdx.x / ktiles, kt = blockIdx.x % ktiles;
  const float* p = src + (size_t)(nt * 16 + li) * K + kt * 32 + half * 8;
  v16bf v = ldrow_bf16(p);
  *(v16u*)(dst + ((size_t)blockIdx.x * 32 + lane) * 16) =
      __builtin_bit_cast(v16u, v);
}

// ---------------- keys^T = Watt^T @ user_rel^T + batt, emitted A-layout ----------------
__global__ void ncf_keys(const float* __restrict__ user_rel, const ushort_t* __restrict__ wattA,
                         const float* __restrict__ batt, ushort_t* __restrict__ keysA) {
  int lane = threadIdx.x, half = lane >> 4, li = lane & 15;
  int u0 = blockIdx.x * 16;
  const float* row = user_rel + (size_t)(u0 + li) * 64;
  v16bf bu0 = ldrow_bf16(row + half * 8);        // k = 0..31
  v16bf bu1 = ldrow_bf16(row + 32 + half * 8);   // k = 32..63
#pragma unroll
  for (int t = 0; t < 2; ++t) {   // output d-chunk pair (2t, 2t+1)
    v8f acc0 = {0.f,0.f,0.f,0.f,0.f,0.f,0.f,0.f};
    v8f acc1 = {0.f,0.f,0.f,0.f,0.f,0.f,0.f,0.f};
    acc0 = wmma_bf16(ld16g(wattA + ((size_t)(0*4 + 2*t)     * 32 + lane) * 16), bu0, acc0);
    acc0 = wmma_bf16(ld16g(wattA + ((size_t)(1*4 + 2*t)     * 32 + lane) * 16), bu1, acc0);
    acc1 = wmma_bf16(ld16g(wattA + ((size_t)(0*4 + 2*t + 1) * 32 + lane) * 16), bu0, acc1);
    acc1 = wmma_bf16(ld16g(wattA + ((size_t)(1*4 + 2*t + 1) * 32 + lane) * 16), bu1, acc1);
    // C-layout halves pair directly into an A-layout tile (K split matches M split)
    union { __bf16 h[16]; v16u v; } o;
    int d0 = 2 * t * 16 + half * 8;
#pragma unroll
    for (int r = 0; r < 8; ++r) {
      o.h[r]     = (__bf16)(acc0[r] + batt[d0 + r]);
      o.h[8 + r] = (__bf16)(acc1[r] + batt[d0 + 16 + r]);
    }
    *(v16u*)(keysA + ((size_t)(blockIdx.x * 2 + t) * 32 + lane) * 16) = o.v;
  }
}

// ---------------- fused relation kernel (no LDS, register-chained) ----------------
// items_relation[i,:] = sum_u sigmoid(item_rel[i]·keys[u]) * mask[u,i] * user_rel[u,:]
__global__ void ncf_relation(const float* __restrict__ mask,
                             const ushort_t* __restrict__ itemB,
                             const ushort_t* __restrict__ keysA,
                             const ushort_t* __restrict__ urelB,
                             float* __restrict__ itemsrel) {
  int lane = threadIdx.x, half = lane >> 4, li = lane & 15;
  int i0 = blockIdx.x * 16;

  v16bf bi0 = ld16g(itemB + ((size_t)(blockIdx.x * 2 + 0) * 32 + lane) * 16);
  v16bf bi1 = ld16g(itemB + ((size_t)(blockIdx.x * 2 + 1) * 32 + lane) * 16);

  v8f accR[4];
#pragma unroll
  for (int dc = 0; dc < 4; ++dc)
    accR[dc] = (v8f){0.f,0.f,0.f,0.f,0.f,0.f,0.f,0.f};

  union { __bf16 h[16]; v16bf v; } a2u;   // masked-score A tile, built across 2 u-tiles
  const int UT = U_N / 16;                // 375
  for (int ut = 0; ut < UT; ++ut) {
    // S^T tile: M = 16 users, N = 16 items (this block's tile)
    v8f s = {0.f,0.f,0.f,0.f,0.f,0.f,0.f,0.f};
    s = wmma_bf16(ld16g(keysA + ((size_t)(ut * 2 + 0) * 32 + lane) * 16), bi0, s);
    s = wmma_bf16(ld16g(keysA + ((size_t)(ut * 2 + 1) * 32 + lane) * 16), bi1, s);

    // element r: (u = ut*16 + half*8 + r, i = i0 + li); coalesced over li
    const float* mcol = mask + (size_t)(ut * 16 + half * 8) * I_N + i0 + li;
    if (ut + 1 < UT)
      __builtin_prefetch(mcol + (size_t)16 * I_N, 0, 1);
    int eo = (ut & 1) * 8;
#pragma unroll
    for (int r = 0; r < 8; ++r) {
      float m = mcol[(size_t)r * I_N];
      float q = 1.0f / (1.0f + __expf(-s[r]));
      a2u.h[eo + r] = (__bf16)(q * m);    // C-layout == A-layout chaining
    }
    bool lastOdd = (ut == UT - 1) && ((ut & 1) == 0);
    if (lastOdd) {
#pragma unroll
      for (int r = 0; r < 8; ++r) a2u.h[8 + r] = (__bf16)0.f;
    }
    if ((ut & 1) == 1 || lastOdd) {
      int ustep = ut >> 1;
#pragma unroll
      for (int dc = 0; dc < 4; ++dc)
        accR[dc] = wmma_bf16(a2u.v,
                             ld16g(urelB + ((size_t)(ustep * 4 + dc) * 32 + lane) * 16),
                             accR[dc]);
    }
  }
#pragma unroll
  for (int dc = 0; dc < 4; ++dc)
#pragma unroll
    for (int r = 0; r < 8; ++r)
      itemsrel[(size_t)(i0 + half * 8 + r) * 64 + dc * 16 + li] = accR[dc][r];
}

// ---------------- fused MLP + GMF + predict (register-resident tower) ----------------
__global__ void ncf_mlp(const int* __restrict__ user, const int* __restrict__ item,
                        const float* __restrict__ eu_gmf, const float* __restrict__ ei_gmf,
                        const float* __restrict__ eu_mlp, const float* __restrict__ ei_mlp,
                        const float* __restrict__ b1, const float* __restrict__ b2,
                        const float* __restrict__ b3, const float* __restrict__ Wp,
                        const float* __restrict__ bp,
                        const ushort_t* __restrict__ w1A, const ushort_t* __restrict__ w2A,
                        const ushort_t* __restrict__ w3A, float* __restrict__ out) {
  int lane = threadIdx.x, half = lane >> 4, li = lane & 15;
  int b0 = blockIdx.x * 16;
  int uid = user[b0 + li], iid = item[b0 + li];
  const float* xu = eu_mlp + (size_t)uid * DM_N;
  const float* xi = ei_mlp + (size_t)iid * DM_N;

  // ---- layer 1: h1^T = relu(W1^T x^T + b1), 256 feats in 8 B-ready v16bf ----
  v16bf h1[8];
#pragma unroll
  for (int p = 0; p < 2; ++p) {
    v8f acc[8];
#pragma unroll
    for (int c = 0; c < 8; ++c) acc[c] = (v8f){0.f,0.f,0.f,0.f,0.f,0.f,0.f,0.f};
    for (int ks = 0; ks < 16; ++ks) {
      int k0 = ks * 32 + half * 8;
      const float* src = (k0 < DM_N) ? (xu + k0) : (xi + (k0 - DM_N));
      v16bf bx = ldrow_bf16(src);
#pragma unroll
      for (int c = 0; c < 8; ++c)
        acc[c] = wmma_bf16(ld16g(w1A + ((size_t)(ks * 16 + p * 8 + c) * 32 + lane) * 16),
                           bx, acc[c]);
    }
#pragma unroll
    for (int jj = 0; jj < 4; ++jj)
      h1[p * 4 + jj] = fuse_pair(acc[2 * jj], acc[2 * jj + 1], b1,
                                 (p * 8 + 2 * jj) * 16 + half * 8, half);
  }

  // ---- layer 2: 256 -> 128, B operands straight from h1 registers ----
  v16bf h2[4];
#pragma unroll
  for (int j = 0; j < 4; ++j) {
    v8f a0 = {0.f,0.f,0.f,0.f,0.f,0.f,0.f,0.f};
    v8f a1 = {0.f,0.f,0.f,0.f,0.f,0.f,0.f,0.f};
#pragma unroll
    for (int ks = 0; ks < 8; ++ks) {
      a0 = wmma_bf16(ld16g(w2A + ((size_t)(ks * 8 + 2*j)     * 32 + lane) * 16), h1[ks], a0);
      a1 = wmma_bf16(ld16g(w2A + ((size_t)(ks * 8 + 2*j + 1) * 32 + lane) * 16), h1[ks], a1);
    }
    h2[j] = fuse_pair(a0, a1, b2, 2 * j * 16 + half * 8, half);
  }

  // ---- layer 3 + predict head, fully in registers ----
  float psum = 0.f;
#pragma unroll
  for (int j = 0; j < 2; ++j) {
    v8f a0 = {0.f,0.f,0.f,0.f,0.f,0.f,0.f,0.f};
    v8f a1 = {0.f,0.f,0.f,0.f,0.f,0.f,0.f,0.f};
#pragma unroll
    for (int ks = 0; ks < 4; ++ks) {
      a0 = wmma_bf16(ld16g(w3A + ((size_t)(ks * 4 + 2*j)     * 32 + lane) * 16), h2[ks], a0);
      a1 = wmma_bf16(ld16g(w3A + ((size_t)(ks * 4 + 2*j + 1) * 32 + lane) * 16), h2[ks], a1);
    }
#pragma unroll
    for (int r = 0; r < 8; ++r) {
      int n0 = 2 * j * 16 + half * 8 + r;
      int n1 = n0 + 16;
      psum += fmaxf(a0[r] + b3[n0], 0.f) * Wp[64 + n0];
      psum += fmaxf(a1[r] + b3[n1], 0.f) * Wp[64 + n1];
    }
  }
  // GMF contribution: this lane's half of d
  {
    const float* gu = eu_gmf + (size_t)uid * D_N + half * 32;
    const float* gi = ei_gmf + (size_t)iid * D_N + half * 32;
    const float* wp = Wp + half * 32;
#pragma unroll
    for (int d = 0; d < 32; d += 4) {
      float4 a = *(const float4*)(gu + d);
      float4 b = *(const float4*)(gi + d);
      float4 w = *(const float4*)(wp + d);
      psum += a.x * b.x * w.x + a.y * b.y * w.y + a.z * b.z * w.z + a.w * b.w * w.w;
    }
  }
  float tot = psum + __shfl_xor(psum, 16, 32);
  if (half == 0) out[b0 + li] = tot + bp[0];
}

// ---------------- batch relation scores ----------------
__global__ void ncf_relscore(const int* __restrict__ user, const int* __restrict__ item,
                             const float* __restrict__ user_rel, const float* __restrict__ item_rel,
                             const float* __restrict__ itemsrel, float* __restrict__ out) {
  int b = blockIdx.x * blockDim.x + threadIdx.x;
  if (b >= B_N) return;
  int u = user[b], it = item[b];
  const float4* ur = (const float4*)(user_rel + (size_t)u * 64);
  const float4* ir = (const float4*)(item_rel + (size_t)it * 64);
  const float4* rr = (const float4*)(itemsrel + (size_t)it * 64);
  float s = 0.f;
#pragma unroll
  for (int q = 0; q < 16; ++q) {
    float4 a = ur[q], c = ir[q], r = rr[q];
    s += (a.x + r.x) * c.x + (a.y + r.y) * c.y + (a.z + r.z) * c.z + (a.w + r.w) * c.w;
  }
  out[B_N + b] = s;
}

// ---------------- host launcher ----------------
extern "C" void kernel_launch(void* const* d_in, const int* in_sizes, int n_in,
                              void* d_out, int out_size, void* d_ws, size_t ws_size,
                              hipStream_t stream) {
  const int*   user     = (const int*)  d_in[0];
  const int*   item     = (const int*)  d_in[1];
  const float* mask     = (const float*)d_in[2];
  const float* eu_gmf   = (const float*)d_in[3];
  const float* ei_gmf   = (const float*)d_in[4];
  const float* eu_mlp   = (const float*)d_in[5];
  const float* ei_mlp   = (const float*)d_in[6];
  const float* W1       = (const float*)d_in[7];
  const float* b1       = (const float*)d_in[8];
  const float* W2       = (const float*)d_in[9];
  const float* b2       = (const float*)d_in[10];
  const float* W3       = (const float*)d_in[11];
  const float* b3       = (const float*)d_in[12];
  const float* Wp       = (const float*)d_in[13];
  const float* bp       = (const float*)d_in[14];
  const float* item_rel = (const float*)d_in[15];
  const float* user_rel = (const float*)d_in[16];
  const float* Watt     = (const float*)d_in[17];
  const float* batt     = (const float*)d_in[18];
  float* out = (float*)d_out;

  // workspace layout (1KB tiles, 256B-aligned offsets)
  char* ws = (char*)d_ws;
  ushort_t* keysA = (ushort_t*)(ws + 0);          // 375*2 tiles = 768000
  ushort_t* itemB = (ushort_t*)(ws + 768000);     // 750*2 tiles = 1536000
  ushort_t* urelB = (ushort_t*)(ws + 2304000);    // 188*4 tiles = 770048
  ushort_t* w1A   = (ushort_t*)(ws + 3074048);    // 16*16      = 262144
  ushort_t* w2A   = (ushort_t*)(ws + 3336192);    // 8*8        = 65536
  ushort_t* w3A   = (ushort_t*)(ws + 3401728);    // 4*4        = 16384
  ushort_t* wattA = (ushort_t*)(ws + 3418112);    // 2*4        = 8192
  float*    irel  = (float*)   (ws + 3426304);    // 12000*64*4 = 3072000

  dim3 wave(32);
  ncf_packBT <<<dim3(750 * 2), wave, 0, stream>>>(item_rel, itemB, I_N, D_N);
  ncf_packBop<<<dim3(188 * 4), wave, 0, stream>>>(user_rel, urelB, U_N, D_N);
  ncf_packAT <<<dim3(16 * 16), wave, 0, stream>>>(W1, w1A, 2 * DM_N, DM_N);
  ncf_packAT <<<dim3(8 * 8),   wave, 0, stream>>>(W2, w2A, DM_N, DM_N / 2);
  ncf_packAT <<<dim3(4 * 4),   wave, 0, stream>>>(W3, w3A, DM_N / 2, D_N);
  ncf_packAT <<<dim3(2 * 4),   wave, 0, stream>>>(Watt, wattA, D_N, D_N);

  ncf_keys    <<<dim3(U_N / 16), wave, 0, stream>>>(user_rel, wattA, batt, keysA);
  ncf_relation<<<dim3(I_N / 16), wave, 0, stream>>>(mask, itemB, keysA, urelB, irel);
  ncf_mlp     <<<dim3(B_N / 16), wave, 0, stream>>>(user, item, eu_gmf, ei_gmf, eu_mlp, ei_mlp,
                                                    b1, b2, b3, Wp, bp, w1A, w2A, w3A, out);
  ncf_relscore<<<dim3(B_N / 256), dim3(256), 0, stream>>>(user, item, user_rel, item_rel,
                                                          irel, out);
}
